// MoEExpertsLoRA_223338300204
// MI455X (gfx1250) — matrix operand order repaired and checked by reading, not compiled, not run on hardware
//
#include <hip/hip_runtime.h>
#include <hip/hip_bf16.h>
#include <stdint.h>

// ---------------------------------------------------------------------------
// MoE + LoRA forward for MI455X (gfx1250), wave32, WMMA bf16x2 split precision.
//
//   k0: zero d_out + per-expert counters
//   k1: split fp32 weights -> (bf16 hi, bf16 lo), transposed to [N][K] (x3)
//   k2: routing gather: per-expert token lists + routing weights
//   k3: fused per-(expert, 16-token-tile) kernel:
//         TDM gather-mode tensor_load_to_lds pulls the 16 routed token rows
//         of X (fp32) into LDS in one DMA (TENSORcnt / s_wait_tensorcnt);
//         stage1: g/u = X@W (bf16x2, 3-term WMMA) + rank-8 LoRA, SiLU*u -> h
//                 kept in 320KB WGP LDS as bf16 hi/lo (contiguous frag layout);
//         stage2: y = h@Wd (+LoRA), routing-weighted global_atomic_add_f32.
// ---------------------------------------------------------------------------

typedef __bf16 bf16_t;
typedef __attribute__((ext_vector_type(16))) __bf16 v16bf;
typedef __attribute__((ext_vector_type(8)))  float  v8f;
typedef __attribute__((ext_vector_type(4)))  unsigned int u32x4;
typedef __attribute__((ext_vector_type(8)))  int   i32x8;
typedef __attribute__((ext_vector_type(4)))  int   i32x4;

#if defined(__has_builtin)
#if __has_builtin(__builtin_amdgcn_tensor_load_to_lds) && __has_builtin(__builtin_amdgcn_s_wait_tensorcnt)
#define HAVE_TDM 1
#endif
#endif

#define E_    8
#define H_    1024
#define I_    2816
#define R_    8
#define T_    4096
#define TOPK_ 2
#define CAP_  (T_ * TOPK_)   // worst-case rows per expert
#define SCALING_ 2.0f
#define TM    16             // token rows per block tile (= TDM gather row limit)

// LDS byte offsets (dynamic smem layout)
#define XH_OFF    0
#define XL_OFF    (TM * H_ * 2)
#define HH_OFF    (2 * TM * H_ * 2)              // 65536: h hi; also fp32 X stage
#define HL_OFF    (HH_OFF + TM * I_ * 2)
#define FSCR_OFF  (HL_OFF + TM * I_ * 2)         // xaG/xaU/haD/tok/tw

// Identity layout of a 32-element K-chunk: one lane's 16-element bf16 WMMA
// A-fragment (lanes 0-15: K = 0..15 ; lanes 16-31: K = 16..31)
// is CONTIGUOUS in LDS -> one 32-byte ds load.
__device__ __forceinline__ int perm32(int k) {
  return k;
}

// --------------------------- k0: zero ---------------------------------------
__global__ void k0_zero(float* __restrict__ out, int* __restrict__ cnt, long n) {
  long i = (long)blockIdx.x * blockDim.x + threadIdx.x;
  if (i < n)  out[i] = 0.0f;
  if (i < E_) cnt[i] = 0;
}

// --------------- k1: fp32 -> bf16 hi/lo, transpose [E][K][N] -> [E][N][K] ---
__global__ void k1_split_w(const float* __restrict__ src,
                           bf16_t* __restrict__ hi, bf16_t* __restrict__ lo,
                           int K, int N, long total) {
  long i = (long)blockIdx.x * blockDim.x + threadIdx.x;
  if (i >= total) return;
  long kn  = (long)K * N;
  int  e   = (int)(i / kn);
  long rem = i - (long)e * kn;
  int  k   = (int)(rem / N);
  int  n   = (int)(rem - (long)k * N);
  float w  = src[i];
  bf16_t h = (bf16_t)w;                 // RNE high part
  bf16_t l = (bf16_t)(w - (float)h);    // residual (exact in fp32)
  long o   = ((long)e * N + n) * (long)K + k;
  hi[o] = h; lo[o] = l;
}

// --------------------------- k2: routing gather -----------------------------
__global__ void k2_route(const float* __restrict__ rw, const int* __restrict__ sel,
                         int* __restrict__ cnt, int* __restrict__ tIdx,
                         float* __restrict__ tW) {
  int i = blockIdx.x * blockDim.x + threadIdx.x;
  if (i >= T_ * TOPK_) return;
  int e = sel[i] & (E_ - 1);
  int p = atomicAdd(&cnt[e], 1);
  tIdx[e * CAP_ + p] = i / TOPK_;       // token id
  tW  [e * CAP_ + p] = rw[i];           // routing weight
}

// ----------------- k3: fused gate/up/SiLU/down per (expert, tile) -----------
__global__ __launch_bounds__(256)
void k3_moe_lora(const float* __restrict__ X,
                 const float* __restrict__ gateA, const float* __restrict__ gateB,
                 const float* __restrict__ upA,   const float* __restrict__ upB,
                 const float* __restrict__ downA, const float* __restrict__ downB,
                 const bf16_t* __restrict__ gHi, const bf16_t* __restrict__ gLo,
                 const bf16_t* __restrict__ uHi, const bf16_t* __restrict__ uLo,
                 const bf16_t* __restrict__ dHi, const bf16_t* __restrict__ dLo,
                 const int* __restrict__ cnt, const int* __restrict__ tIdx,
                 const float* __restrict__ tW, float* __restrict__ out) {
  const int e     = blockIdx.y;
  const int tile  = blockIdx.x;
  const int nrows = cnt[e];
  if (tile * TM >= nrows) return;       // block-uniform early exit

  extern __shared__ __align__(32) char smem[];
  bf16_t* XH   = (bf16_t*)(smem + XH_OFF);   // [TM][H_]  contiguous A layout
  bf16_t* XL   = (bf16_t*)(smem + XL_OFF);
  bf16_t* HH   = (bf16_t*)(smem + HH_OFF);   // [TM][I_]  contiguous A layout
  bf16_t* HL   = (bf16_t*)(smem + HL_OFF);
  float*  stage= (float*)(smem + HH_OFF);    // fp32 X tile (dead before h lives)
  float*  xaG  = (float*)(smem + FSCR_OFF);  // [TM][R_]  (pre-scaled by SCALING)
  float*  xaU  = xaG + TM * R_;
  float*  haD  = xaU + TM * R_;
  int*    tokS = (int*)(haD + TM * R_);      // [TM]
  float*  twS  = (float*)(tokS + TM);        // [TM]

  const int tid  = threadIdx.x;
  const int lane = tid & 31;
  const int wave = tid >> 5;            // 8 waves
  const int ncol = lane & 15;           // WMMA N column / A row
  const int hsel = lane >> 4;           // lane half
  const int ko   = hsel * 16;           // per-lane K offset inside 32-chunk

  // ---- phase A: row bookkeeping
  if (tid < TM) {
    int row = tile * TM + tid;
    if (row < nrows) { tokS[tid] = tIdx[e * CAP_ + row]; twS[tid] = tW[e * CAP_ + row]; }
    else             { tokS[tid] = -1;                   twS[tid] = 0.0f; }
  }
  __syncthreads();

#ifdef HAVE_TDM
  // ---- phase B0: TDM gather-mode DMA — fetch the 16 routed token rows of X
  // (fp32, 16 x 1024) straight into LDS with one tensor_load_to_lds.
  if (wave == 0) {
    unsigned long long xb = (unsigned long long)(uintptr_t)X;
    // D# group0: count=1 | gather_mode (16-bit idx) | lds_addr | global_addr | type=2
    u32x4 g0 = { 0x80000001u,
                 (unsigned int)HH_OFF,
                 (unsigned int)(xb & 0xFFFFFFFFu),
                 (unsigned int)((xb >> 32) & 0x01FFFFFFu) | (2u << 30) };
    // D# group1: data_size=4B; tensor_dim0=H_ (bits79:48); tensor_dim1=T_
    // (bits111:80); tile_dim0=H_ (bits127:112); tile_dim1=16 idx; dim0_stride=H_
    i32x8 g1 = { 0x00020000,
                 (int)(H_ << 16),
                 (int)(T_ << 16),
                 (int)(H_ << 16),
                 16,
                 H_,
                 0, 0 };
    int id[16];
#pragma unroll
    for (int m = 0; m < 16; ++m) {
      int t = tokS[m];
      id[m] = __builtin_amdgcn_readfirstlane(t < 0 ? 0 : t);
    }
    // D# groups 2-3: sixteen 16-bit row indices (the gathered token ids)
    i32x4 g2 = { id[0] | (id[1] << 16), id[2] | (id[3] << 16),
                 id[4] | (id[5] << 16), id[6] | (id[7] << 16) };
    i32x4 g3 = { id[8]  | (id[9]  << 16), id[10] | (id[11] << 16),
                 id[12] | (id[13] << 16), id[14] | (id[15] << 16) };
    i32x8 gpad = { 0, 0, 0, 0, 0, 0, 0, 0 };   // 6-arg form (clang-23 lane)
    __builtin_amdgcn_tensor_load_to_lds(g0, g1, g2, g3, gpad, 0);
    __builtin_amdgcn_s_wait_tensorcnt(0);
  }
  __syncthreads();

  // ---- phase B1: split staged fp32 X -> bf16 hi/lo, contiguous frag layout
  for (int i = tid; i < TM * H_; i += 256) {
    int m = i >> 10;                    // / H_
    int k = i & (H_ - 1);
    float x = (tokS[m] >= 0) ? stage[i] : 0.0f;
    bf16_t xh = (bf16_t)x;
    bf16_t xl = (bf16_t)(x - (float)xh);
    int pos = m * H_ + (k & ~31) + perm32(k & 31);
    XH[pos] = xh; XL[pos] = xl;
  }
#else
  // ---- fallback: direct global gather of X
  for (int i = tid; i < TM * H_; i += 256) {
    int m = i >> 10;
    int k = i & (H_ - 1);
    int t = tokS[m];
    float x = (t >= 0) ? X[(size_t)t * H_ + k] : 0.0f;
    bf16_t xh = (bf16_t)x;
    bf16_t xl = (bf16_t)(x - (float)xh);
    int pos = m * H_ + (k & ~31) + perm32(k & 31);
    XH[pos] = xh; XL[pos] = xl;
  }
#endif
  __syncthreads();

  // ---- phase C: LoRA A-side  xa = SCALING * (X @ A)   [TM x R], rank 8
  if (tid < 2 * TM * R_) {
    const float* A  = (tid < TM * R_) ? gateA : upA;
    float*      dst = (tid < TM * R_) ? xaG   : xaU;
    int j = tid & (TM * R_ - 1);
    int m = j >> 3, r = j & 7;
    float acc = 0.f;
    for (int k = 0; k < H_; ++k) {
      int pos = m * H_ + (k & ~31) + perm32(k & 31);
      acc += ((float)XH[pos] + (float)XL[pos]) * A[((size_t)e * H_ + k) * R_ + r];
    }
    dst[j] = acc * SCALING_;
  }
  __syncthreads();

  // ---- phase D (stage 1): g/u = X@W (bf16x2, 3-term), SiLU, h -> LDS
  for (int tn = wave; tn < I_ / 16; tn += 8) {
    const int n = tn * 16 + ncol;
    const bf16_t* bgh = gHi + ((size_t)e * I_ + n) * H_;  // transposed [N][K]
    const bf16_t* bgl = gLo + ((size_t)e * I_ + n) * H_;
    const bf16_t* buh = uHi + ((size_t)e * I_ + n) * H_;
    const bf16_t* bul = uLo + ((size_t)e * I_ + n) * H_;
    v8f gacc = {0,0,0,0,0,0,0,0};
    v8f uacc = {0,0,0,0,0,0,0,0};
    for (int kk = 0; kk < H_; kk += 32) {
      __builtin_prefetch(bgh + kk + 256, 0, 1);           // global_prefetch_b8
      __builtin_prefetch(buh + kk + 256, 0, 1);
      v16bf a_h = *(const v16bf*)(XH + ncol * H_ + kk + ko);
      v16bf a_l = *(const v16bf*)(XL + ncol * H_ + kk + ko);
      v16bf wgh = *(const v16bf*)(bgh + kk + ko);
      v16bf wgl = *(const v16bf*)(bgl + kk + ko);
      gacc = __builtin_amdgcn_wmma_f32_16x16x32_bf16(false, a_h, false, wgh, (short)0, gacc, false, false);
      gacc = __builtin_amdgcn_wmma_f32_16x16x32_bf16(false, a_h, false, wgl, (short)0, gacc, false, false);
      gacc = __builtin_amdgcn_wmma_f32_16x16x32_bf16(false, a_l, false, wgh, (short)0, gacc, false, false);
      v16bf wuh = *(const v16bf*)(buh + kk + ko);
      v16bf wul = *(const v16bf*)(bul + kk + ko);
      uacc = __builtin_amdgcn_wmma_f32_16x16x32_bf16(false, a_h, false, wuh, (short)0, uacc, false, false);
      uacc = __builtin_amdgcn_wmma_f32_16x16x32_bf16(false, a_h, false, wul, (short)0, uacc, false, false);
      uacc = __builtin_amdgcn_wmma_f32_16x16x32_bf16(false, a_l, false, wuh, (short)0, uacc, false, false);
    }
    // epilogue: rank-8 LoRA B-side, SiLU, split-store h
#pragma unroll
    for (int v = 0; v < 8; ++v) {
      int m = v + hsel * 8;             // C/D layout: lanes16-31 hold M=v+8
      float g = gacc[v], u = uacc[v];
#pragma unroll
      for (int r = 0; r < R_; ++r) {
        g += xaG[m * R_ + r] * gateB[((size_t)e * R_ + r) * I_ + n];
        u += xaU[m * R_ + r] * upB  [((size_t)e * R_ + r) * I_ + n];
      }
      float hval = (g / (1.0f + __expf(-g))) * u;         // SiLU(g) * u
      bf16_t hh = (bf16_t)hval;
      bf16_t hl = (bf16_t)(hval - (float)hh);
      int pos = m * I_ + (n & ~31) + perm32(n & 31);
      HH[pos] = hh; HL[pos] = hl;
    }
  }
  if (tid < TM * R_) haD[tid] = 0.f;
  __syncthreads();

  // ---- phase E: LoRA A-side for down  ha = SCALING * (h @ downA)
  // all 256 threads: 2-way K split per (m,r), ds_add_f32 reduction
  {
    int j = tid & (TM * R_ - 1);
    int m = j >> 3, r = j & 7;
    int k0 = (tid >> 7) * (I_ / 2);
    int k1 = k0 + I_ / 2;
    float acc = 0.f;
    for (int k = k0; k < k1; ++k) {
      int pos = m * I_ + (k & ~31) + perm32(k & 31);
      acc += ((float)HH[pos] + (float)HL[pos]) * downA[((size_t)e * I_ + k) * R_ + r];
    }
    atomicAdd(&haD[j], acc * SCALING_);
  }
  __syncthreads();

  // ---- phase F (stage 2): y = h @ Wd (+LoRA), weighted scatter-add
  for (int tn = wave; tn < H_ / 16; tn += 8) {
    const int n = tn * 16 + ncol;
    const bf16_t* bdh = dHi + ((size_t)e * H_ + n) * I_;  // transposed [N][K]
    const bf16_t* bdl = dLo + ((size_t)e * H_ + n) * I_;
    v8f acc = {0,0,0,0,0,0,0,0};
    for (int kk = 0; kk < I_; kk += 32) {
      __builtin_prefetch(bdh + kk + 256, 0, 1);
      v16bf a_h = *(const v16bf*)(HH + ncol * I_ + kk + ko);
      v16bf a_l = *(const v16bf*)(HL + ncol * I_ + kk + ko);
      v16bf wdh = *(const v16bf*)(bdh + kk + ko);
      v16bf wdl = *(const v16bf*)(bdl + kk + ko);
      acc = __builtin_amdgcn_wmma_f32_16x16x32_bf16(false, a_h, false, wdh, (short)0, acc, false, false);
      acc = __builtin_amdgcn_wmma_f32_16x16x32_bf16(false, a_h, false, wdl, (short)0, acc, false, false);
      acc = __builtin_amdgcn_wmma_f32_16x16x32_bf16(false, a_l, false, wdh, (short)0, acc, false, false);
    }
#pragma unroll
    for (int v = 0; v < 8; ++v) {
      int m = v + hsel * 8;
      float y = acc[v];
#pragma unroll
      for (int r = 0; r < R_; ++r)
        y += haD[m * R_ + r] * downB[((size_t)e * R_ + r) * H_ + n];
      int t = tokS[m];
      if (t >= 0) atomicAdd(&out[(size_t)t * H_ + n], twS[m] * y);
    }
  }
}

// ---------------------------------------------------------------------------
extern "C" void kernel_launch(void* const* d_in, const int* in_sizes, int n_in,
                              void* d_out, int out_size, void* d_ws, size_t ws_size,
                              hipStream_t stream) {
  const float* X     = (const float*)d_in[0];
  const float* rw    = (const float*)d_in[1];
  const int*   sel   = (const int*)  d_in[2];
  const float* gateP = (const float*)d_in[3];
  const float* upP   = (const float*)d_in[4];
  const float* downP = (const float*)d_in[5];
  const float* gateA = (const float*)d_in[6];
  const float* gateB = (const float*)d_in[7];
  const float* upA   = (const float*)d_in[8];
  const float* upB   = (const float*)d_in[9];
  const float* downA = (const float*)d_in[10];
  const float* downB = (const float*)d_in[11];
  float* out = (float*)d_out;

  // workspace carve-up (~278 MB: 6 transposed bf16 weight planes + routing)
  char*  ws  = (char*)d_ws;
  int*   cnt = (int*)ws;
  size_t off = 256;
  int*   tIdx = (int*)  (ws + off); off += (size_t)E_ * CAP_ * 4;
  float* tW   = (float*)(ws + off); off += (size_t)E_ * CAP_ * 4;
  off = (off + 255) & ~(size_t)255;
  const size_t WB = (size_t)E_ * H_ * I_ * sizeof(bf16_t);  // bytes per plane
  bf16_t* gHi = (bf16_t*)(ws + off); off += WB;
  bf16_t* gLo = (bf16_t*)(ws + off); off += WB;
  bf16_t* uHi = (bf16_t*)(ws + off); off += WB;
  bf16_t* uLo = (bf16_t*)(ws + off); off += WB;
  bf16_t* dHi = (bf16_t*)(ws + off); off += WB;
  bf16_t* dLo = (bf16_t*)(ws + off); off += WB;

  // k0: zero output + expert counters
  long outN = (long)T_ * H_;
  k0_zero<<<(int)((outN + 255) / 256), 256, 0, stream>>>(out, cnt, outN);

  // k1: split/transposed bf16 weight planes (streaming, ~memory-bound)
  long tot = (long)E_ * H_ * I_;
  int  cb  = (int)((tot + 255) / 256);
  k1_split_w<<<cb, 256, 0, stream>>>(gateP, gHi, gLo, H_, I_, tot);
  k1_split_w<<<cb, 256, 0, stream>>>(upP,   uHi, uLo, H_, I_, tot);
  k1_split_w<<<cb, 256, 0, stream>>>(downP, dHi, dLo, I_, H_, tot);

  // k2: routing gather
  k2_route<<<(T_ * TOPK_ + 255) / 256, 256, 0, stream>>>(rw, sel, cnt, tIdx, tW);

  // k3: fused MoE-LoRA; dynamic LDS ~242KB (fits MI455X 320KB WGP LDS)
  size_t smem = (size_t)FSCR_OFF
              + 3 * TM * R_ * sizeof(float) + TM * (sizeof(int) + sizeof(float));
  k3_moe_lora<<<dim3(CAP_ / TM, E_), 256, smem, stream>>>(
      X, gateA, gateB, upA, upB, downA, downB,
      gHi, gLo, uHi, uLo, dHi, dLo, cnt, tIdx, tW, out);
}
